// MixtureOfExperts_58265526337941
// MI455X (gfx1250) — matrix operand order repaired and checked by reading, not compiled, not run on hardware
//
#include <hip/hip_runtime.h>
#include <hip/hip_bf16.h>
#include <stddef.h>

// Problem constants (from reference)
#define B_SZ   16384
#define D_SZ   512
#define E_SZ   16
#define H_SZ   1024
#define H2_SZ  512
#define GD_SZ  256
#define OUT_SZ 2

#define BM   64      // batch rows per workgroup
#define PAD  8       // LDS row padding (bf16 elements) -> 16B, de-conflicts banks

typedef __attribute__((ext_vector_type(8)))  __bf16 v8bf;
typedef __attribute__((ext_vector_type(16))) __bf16 v16bf;
typedef __attribute__((ext_vector_type(8)))  float  v8f;

__device__ __forceinline__ unsigned short f2bf(float f) {
  unsigned u = __float_as_uint(f);
  u += 0x7fffu + ((u >> 16) & 1u);          // round-to-nearest-even
  return (unsigned short)(u >> 16);
}
__device__ __forceinline__ float bf2f(unsigned short s) {
  return __uint_as_float(((unsigned)s) << 16);
}

// ---- WMMA fragment helpers (wave32, v_wmma_f32_16x16x32_bf16) -------------
// A fragment: 16x32 (MxK). Lanes 0-15 row M=lane: K = k0+{0..7}, k0+16+{0..7};
// lanes 16-31 same rows with K offset +8.
__device__ __forceinline__ v16bf frag_a_lds(const unsigned short* base, int ldK,
                                            int rowBase, int k0, int lane) {
  const int half = lane >> 4, m = lane & 15;
  const unsigned short* p = base + (size_t)(rowBase + m) * ldK + k0 + half * 8;
  v8bf lo = *(const v8bf*)p;          // K = k0 + half*8 .. +7
  v8bf hi = *(const v8bf*)(p + 16);   // K = k0 + 16 + half*8 .. +7
  return __builtin_shufflevector(lo, hi, 0,1,2,3,4,5,6,7,8,9,10,11,12,13,14,15);
}

// B fragment: 32x16 (KxN) from transposed weights Wt[n][k] (contiguous K).
// Lanes 0-15 col n=lane: K = k0+{0..7}, k0+16+{0..7}; lanes 16-31 same cols with K offset +8.
__device__ __forceinline__ v16bf frag_b_glb(const unsigned short* wt, int ldK,
                                            int colBase, int k0, int lane) {
  const int half = lane >> 4, n = lane & 15;
  const unsigned short* p = wt + (size_t)(colBase + n) * ldK + k0 + half * 8;
  v8bf lo = *(const v8bf*)p;
  v8bf hi = *(const v8bf*)(p + 16);
  return __builtin_shufflevector(lo, hi, 0,1,2,3,4,5,6,7,8,9,10,11,12,13,14,15);
}

__device__ __forceinline__ v8f wmma_bf16(v16bf a, v16bf b, v8f c) {
  return __builtin_amdgcn_wmma_f32_16x16x32_bf16(false, a, false, b, (short)0, c,
                                                 false, false);
}

// C/D layout: lane -> N = lane%16, VGPR i -> M = i + 8*(lane/16)
__device__ __forceinline__ void store_relu_bf16(unsigned short* dst, int ldK,
                                                int rowBase, int colBase,
                                                const v8f& c, float bias, int lane) {
  const int half = lane >> 4, n = lane & 15;
#pragma unroll
  for (int i = 0; i < 8; i++) {
    float v = c[i] + bias;
    v = v > 0.f ? v : 0.f;
    dst[(size_t)(rowBase + i + 8 * half) * ldK + colBase + n] = f2bf(v);
  }
}

// One 64-row x 32-col macro-tile (4 row-tiles x 2 col-tiles) GEMM pass with
// bias+relu, A from LDS, B from global (transposed weights), out -> LDS bf16.
// Register budget: 64 acc + 32 A + 16 B VGPRs. k-loop deliberately NOT
// unrolled so the compiler cannot hoist the whole A panel into registers.
__device__ __forceinline__ void macro_tile_gemm(
    const unsigned short* __restrict__ sSrc, int ldSrc,       // A: [64][K]
    const unsigned short* __restrict__ wt, int K,             // B: [N][K] global
    const float* __restrict__ bias, int colBase,              // bias[colBase..]
    unsigned short* __restrict__ sDst, int ldDst,             // D: [64][N]
    int lane) {
  v8f acc[4][2] = {};
#pragma unroll 1
  for (int k0 = 0; k0 < K; k0 += 32) {
    v16bf b0 = frag_b_glb(wt, K, colBase, k0, lane);
    v16bf b1 = frag_b_glb(wt, K, colBase + 16, k0, lane);
#pragma unroll
    for (int rt = 0; rt < 4; rt++) {
      v16bf a = frag_a_lds(sSrc, ldSrc, rt * 16, k0, lane);
      acc[rt][0] = wmma_bf16(a, b0, acc[rt][0]);
      acc[rt][1] = wmma_bf16(a, b1, acc[rt][1]);
    }
  }
  float bias0 = bias[colBase + (lane & 15)];
  float bias1 = bias[colBase + 16 + (lane & 15)];
#pragma unroll
  for (int rt = 0; rt < 4; rt++) {
    store_relu_bf16(sDst, ldDst, rt * 16, colBase, acc[rt][0], bias0, lane);
    store_relu_bf16(sDst, ldDst, rt * 16, colBase + 16, acc[rt][1], bias1, lane);
  }
}

// ---- Preprocessing ---------------------------------------------------------
__global__ void cvt_f32_bf16(const float* __restrict__ src,
                             unsigned short* __restrict__ dst, int n) {
  int i = blockIdx.x * 256 + threadIdx.x;
  if (i < n) dst[i] = f2bf(src[i]);
}

// src: [N][R][C] f32  ->  dst: [N][C][R] bf16   (N = gridDim.y slice)
__global__ void transpose_cvt(const float* __restrict__ src,
                              unsigned short* __restrict__ dst, int R, int C) {
  size_t base = (size_t)blockIdx.y * R * C;
  int i = blockIdx.x * 256 + threadIdx.x;
  if (i >= R * C) return;
  int r = i % R, c = i / R;
  dst[base + (size_t)c * R + r] = f2bf(src[base + (size_t)r * C + c]);
}

__global__ void zero_f32(float* __restrict__ p, int n) {
  int i = blockIdx.x * 256 + threadIdx.x;
  if (i < n) p[i] = 0.f;
}

// ---- Gating: gate[B,E] = softmax(relu(relu(x G1 + g1) G2 + g2) G3 + g3) ---
__global__ __launch_bounds__(256) void moe_gating_kernel(
    const unsigned short* __restrict__ xb,
    const unsigned short* __restrict__ G1t,   // [GD][D]
    const unsigned short* __restrict__ G2t,   // [GD][GD]
    const unsigned short* __restrict__ G3t,   // [E][GD]
    const float* __restrict__ g1, const float* __restrict__ g2,
    const float* __restrict__ g3, float* __restrict__ gate) {
  __shared__ alignas(16) unsigned short sX[BM * (D_SZ + PAD)];
  __shared__ alignas(16) unsigned short sH1[BM * (GD_SZ + PAD)];
  __shared__ alignas(16) unsigned short sH2[BM * (GD_SZ + PAD)];
  __shared__ float sLog[BM * E_SZ];

  const int tid = threadIdx.x, lane = tid & 31, wave = tid >> 5;
  const int rowBase = blockIdx.x * BM;

  {  // stage x tile
    const unsigned short* src = xb + (size_t)rowBase * D_SZ;
    for (int j = tid; j < BM * D_SZ / 8; j += 256) {
      int r = (j * 8) / D_SZ, c = (j * 8) % D_SZ;
      *(v8bf*)&sX[r * (D_SZ + PAD) + c] = *(const v8bf*)&src[(size_t)r * D_SZ + c];
    }
  }
  __syncthreads();

  // L1: [64 x 512] @ [512 x 256] ; 8 col-pairs == 8 waves
  macro_tile_gemm(sX, D_SZ + PAD, G1t, D_SZ, g1, wave * 32,
                  sH1, GD_SZ + PAD, lane);
  __syncthreads();

  // L2: [64 x 256] @ [256 x 256]
  macro_tile_gemm(sH1, GD_SZ + PAD, G2t, GD_SZ, g2, wave * 32,
                  sH2, GD_SZ + PAD, lane);
  __syncthreads();

  // L3: [64 x 256] @ [256 x 16] -> logits (f32 in LDS); 4 row-tiles, waves 0-3
  if (wave < 4) {
    const int rt = wave;
    v8f acc = {};
#pragma unroll 1
    for (int k0 = 0; k0 < GD_SZ; k0 += 32) {
      v16bf b = frag_b_glb(G3t, GD_SZ, 0, k0, lane);
      v16bf a = frag_a_lds(sH2, GD_SZ + PAD, rt * 16, k0, lane);
      acc = wmma_bf16(a, b, acc);
    }
    const int half = lane >> 4, n = lane & 15;
    float bias = g3[n];
#pragma unroll
    for (int i = 0; i < 8; i++)
      sLog[(rt * 16 + i + 8 * half) * E_SZ + n] = acc[i] + bias;
  }
  __syncthreads();

  // softmax over E=16 per row
  if (tid < BM) {
    float m = -3.4e38f;
#pragma unroll
    for (int j = 0; j < E_SZ; j++) m = fmaxf(m, sLog[tid * E_SZ + j]);
    float ex[E_SZ], s = 0.f;
#pragma unroll
    for (int j = 0; j < E_SZ; j++) { ex[j] = __expf(sLog[tid * E_SZ + j] - m); s += ex[j]; }
    float inv = 1.f / s;
#pragma unroll
    for (int j = 0; j < E_SZ; j++)
      gate[(size_t)(rowBase + tid) * E_SZ + j] = ex[j] * inv;
  }
}

// ---- Expert pipeline: fused 3-layer MLP per (expert, 64-row tile) ---------
__global__ __launch_bounds__(256) void moe_expert_kernel(
    const unsigned short* __restrict__ xb,     // [B][D] bf16
    const unsigned short* __restrict__ W1t,    // [E][H][D] bf16
    const unsigned short* __restrict__ W2t,    // [E][H2][H] bf16
    const unsigned short* __restrict__ W3t,    // [E][OUT][H2] bf16
    const float* __restrict__ b1, const float* __restrict__ b2,
    const float* __restrict__ b3,
    const float* __restrict__ gate,            // [B][E] f32
    float* __restrict__ out) {                 // [B][OUT] f32 (pre-zeroed)
  __shared__ alignas(16) unsigned short sA[BM * (D_SZ + PAD)];   // x tile, then h2 (H2==D)
  __shared__ alignas(16) unsigned short sH1[BM * (H_SZ + PAD)];

  const int tid = threadIdx.x, lane = tid & 31, wave = tid >> 5;
  const int e = blockIdx.y;
  const int rowBase = blockIdx.x * BM;

  {  // stage x tile (coalesced b128)
    const unsigned short* src = xb + (size_t)rowBase * D_SZ;
    for (int j = tid; j < BM * D_SZ / 8; j += 256) {
      int r = (j * 8) / D_SZ, c = (j * 8) % D_SZ;
      *(v8bf*)&sA[r * (D_SZ + PAD) + c] = *(const v8bf*)&src[(size_t)r * D_SZ + c];
    }
  }
  __syncthreads();

  // Layer 1: h1 = relu(x @ W1[e] + b1[e])   [64 x 512] @ [512 x 1024]
  // 32 col-pairs over 8 waves -> 4 macro-tiles per wave.
  const unsigned short* w1 = W1t + (size_t)e * H_SZ * D_SZ;
  const float* bias1 = b1 + e * H_SZ;
#pragma unroll 1
  for (int ctp = wave; ctp < H_SZ / 32; ctp += 8)
    macro_tile_gemm(sA, D_SZ + PAD, w1, D_SZ, bias1, ctp * 32,
                    sH1, H_SZ + PAD, lane);
  __syncthreads();

  // Layer 2: h2 = relu(h1 @ W2[e] + b2[e])  [64 x 1024] @ [1024 x 512]
  // h2 overwrites the x-tile region (x no longer needed). 16 col-pairs.
  const unsigned short* w2 = W2t + (size_t)e * H2_SZ * H_SZ;
  const float* bias2 = b2 + e * H2_SZ;
#pragma unroll 1
  for (int ctp = wave; ctp < H2_SZ / 32; ctp += 8)
    macro_tile_gemm(sH1, H_SZ + PAD, w2, H_SZ, bias2, ctp * 32,
                    sA, H2_SZ + PAD, lane);
  __syncthreads();

  // Layer 3 + gated combine: out[b,:] += gate[b,e] * (h2 @ W3[e] + b3[e])
  if (tid < BM * OUT_SZ) {
    int r = tid >> 1, o = tid & 1;
    const unsigned short* w3 = W3t + ((size_t)e * OUT_SZ + o) * H2_SZ;
    float acc = b3[e * OUT_SZ + o];
    for (int k = 0; k < H2_SZ; k++)
      acc += bf2f(sA[r * (H2_SZ + PAD) + k]) * bf2f(w3[k]);
    float g = gate[(size_t)(rowBase + r) * E_SZ + e];
    atomicAdd(&out[(size_t)(rowBase + r) * OUT_SZ + o], g * acc);
  }
}

// ---- Host launch -----------------------------------------------------------
extern "C" void kernel_launch(void* const* d_in, const int* in_sizes, int n_in,
                              void* d_out, int out_size, void* d_ws, size_t ws_size,
                              hipStream_t stream) {
  const float* x  = (const float*)d_in[0];
  const float* G1 = (const float*)d_in[1];
  const float* g1 = (const float*)d_in[2];
  const float* G2 = (const float*)d_in[3];
  const float* g2 = (const float*)d_in[4];
  const float* G3 = (const float*)d_in[5];
  const float* g3 = (const float*)d_in[6];
  const float* W1 = (const float*)d_in[7];
  const float* b1 = (const float*)d_in[8];
  const float* W2 = (const float*)d_in[9];
  const float* b2 = (const float*)d_in[10];
  const float* W3 = (const float*)d_in[11];
  const float* b3 = (const float*)d_in[12];
  float* out = (float*)d_out;

  char* ws = (char*)d_ws;
  size_t off = 0;
  auto wsalloc = [&](size_t bytes) -> void* {
    void* p = ws + off;
    off += (bytes + 255) & ~(size_t)255;
    return p;
  };
  unsigned short* xb  = (unsigned short*)wsalloc((size_t)B_SZ * D_SZ * 2);
  unsigned short* W1t = (unsigned short*)wsalloc((size_t)E_SZ * H_SZ * D_SZ * 2);
  unsigned short* W2t = (unsigned short*)wsalloc((size_t)E_SZ * H2_SZ * H_SZ * 2);
  unsigned short* W3t = (unsigned short*)wsalloc((size_t)E_SZ * OUT_SZ * H2_SZ * 2);
  unsigned short* G1t = (unsigned short*)wsalloc((size_t)GD_SZ * D_SZ * 2);
  unsigned short* G2t = (unsigned short*)wsalloc((size_t)GD_SZ * GD_SZ * 2);
  unsigned short* G3t = (unsigned short*)wsalloc((size_t)E_SZ * GD_SZ * 2);
  float* gate = (float*)wsalloc((size_t)B_SZ * E_SZ * 4);

  // Preprocess: f32 -> bf16, weights transposed to [N][K] for WMMA B-frags.
  cvt_f32_bf16<<<(B_SZ * D_SZ) / 256, 256, 0, stream>>>(x, xb, B_SZ * D_SZ);
  transpose_cvt<<<dim3((D_SZ * H_SZ) / 256, E_SZ), 256, 0, stream>>>(W1, W1t, D_SZ, H_SZ);
  transpose_cvt<<<dim3((H_SZ * H2_SZ) / 256, E_SZ), 256, 0, stream>>>(W2, W2t, H_SZ, H2_SZ);
  transpose_cvt<<<dim3((H2_SZ * OUT_SZ + 255) / 256, E_SZ), 256, 0, stream>>>(W3, W3t, H2_SZ, OUT_SZ);
  transpose_cvt<<<dim3((D_SZ * GD_SZ) / 256, 1), 256, 0, stream>>>(G1, G1t, D_SZ, GD_SZ);
  transpose_cvt<<<dim3((GD_SZ * GD_SZ) / 256, 1), 256, 0, stream>>>(G2, G2t, GD_SZ, GD_SZ);
  transpose_cvt<<<dim3((GD_SZ * E_SZ + 255) / 256, 1), 256, 0, stream>>>(G3, G3t, GD_SZ, E_SZ);
  zero_f32<<<(B_SZ * OUT_SZ + 255) / 256, 256, 0, stream>>>(out, B_SZ * OUT_SZ);

  // Gating then fused expert pipeline.
  moe_gating_kernel<<<B_SZ / BM, 256, 0, stream>>>(xb, G1t, G2t, G3t, g1, g2, g3, gate);
  moe_expert_kernel<<<dim3(B_SZ / BM, E_SZ), 256, 0, stream>>>(xb, W1t, W2t, W3t,
                                                               b1, b2, b3, gate, out);
  (void)in_sizes; (void)n_in; (void)out_size; (void)ws_size;
}
